// Attention_51238959841444
// MI455X (gfx1250) — matrix operand order repaired and checked by reading, not compiled, not run on hardware
//
#include <hip/hip_runtime.h>

// MI455X (gfx1250, wave32) windowed attention.
// One 256-thread block (8 waves) per 64-token window; wave == head.
// All matmuls via V_WMMA_F32_16X16X32_F16 (f16 in, f32 accumulate).
// x/c are staged fp32->f16 into LDS exactly once per block; LDS regions are
// retimed: R0 Xh->V^T, R1 Ch->P(lo), R2 Q->P(hi), R3 K->O.

typedef __attribute__((ext_vector_type(16))) _Float16 v16h;
typedef __attribute__((ext_vector_type(8)))  _Float16 v8h;
typedef __attribute__((ext_vector_type(4)))  _Float16 v4h;
typedef __attribute__((ext_vector_type(8)))  float    v8f;
typedef __attribute__((ext_vector_type(4)))  float    v4f;

#define D_MODEL 256
#define N_TOK   64
#define N_HEAD  8
#define D_HEAD  32
#define WIN_ELEMS (N_TOK * D_MODEL)          // 16384 elems per window
#define TAB_ELEMS (225 * N_HEAD)             // (2W-1)^2 * heads

__device__ __forceinline__ v8f wmma_f16(v16h a, v16h b, v8f c) {
  // D = A*B + C, 16x16x32 f16 -> f32
  return __builtin_amdgcn_wmma_f32_16x16x32_f16(false, a, false, b, (short)0, c,
                                                false, false);
}

// A-fragment (16x32 f16) from row-major f16 in LDS.
// ISA layout: lanes 0-15 row M=lane, K = {0..7,16..23}; lanes 16-31 row M=lane-16,
// K = {8..15,24..31}.  Two 16-byte chunks -> 2x ds_load_b128.
__device__ __forceinline__ v16h a_frag_l(const _Float16* base, int ld,
                                         int row0, int k0, int lane) {
  const int r = lane & 15, kb = (lane >> 4) << 3;
  const _Float16* p = base + (row0 + r) * ld + k0 + kb;
  v8h lo = *reinterpret_cast<const v8h*>(p);
  v8h hi = *reinterpret_cast<const v8h*>(p + 16);
  v16h a;
#pragma unroll
  for (int t = 0; t < 8; ++t) { a[t] = lo[t]; a[8 + t] = hi[t]; }
  return a;
}

// B-fragment (32x16 f16) from Bt[n][k] (B stored column-major / W transposed).
// ISA layout: lane&15 = column N; lanes 0-15 hold K = {0..7,16..23}, lanes 16-31
// K = {8..15,24..31} (matches A fragment) -> two 16-byte chunks per lane.
__device__ __forceinline__ v16h b_frag(const _Float16* bt, int ld,
                                       int n0, int k0, int lane) {
  const int n = lane & 15, kh = (lane >> 4) << 3;
  const _Float16* p = bt + (n0 + n) * ld + k0 + kh;
  v8h lo = *reinterpret_cast<const v8h*>(p);
  v8h hi = *reinterpret_cast<const v8h*>(p + 16);
  v16h b;
#pragma unroll
  for (int t = 0; t < 8; ++t) { b[t] = lo[t]; b[8 + t] = hi[t]; }
  return b;
}

// C tile (16x16 f32) -> row-major f16 LDS, with scale.
__device__ __forceinline__ void store_tile_f16(_Float16* dst, int ld, int row0,
                                               int col0, v8f c, float scale,
                                               int lane) {
  const int n = lane & 15, hb = (lane >> 4) << 3;
#pragma unroll
  for (int r = 0; r < 8; ++r)
    dst[(row0 + hb + r) * ld + col0 + n] = (_Float16)(c[r] * scale);
}

// C tile -> transposed f16 LDS ([col][row]); 8 consecutive rows per lane -> 1 b128.
__device__ __forceinline__ void store_tile_f16_T(_Float16* dst, int ldT, int row0,
                                                 int col0, v8f c, int lane) {
  const int n = lane & 15, hb = (lane >> 4) << 3;
  v8h pk;
#pragma unroll
  for (int r = 0; r < 8; ++r) pk[r] = (_Float16)c[r];
  *reinterpret_cast<v8h*>(dst + (col0 + n) * ldT + row0 + hb) = pk;
}

// Per-head projection: dst_head = srcA(64x256 f16, LDS) @ Wt^T cols [n0, n0+32).
template <bool TRANSPOSED>
__device__ __forceinline__ void proj_head(const _Float16* srcA,
                                          const _Float16* __restrict__ wT,
                                          _Float16* dst, int n0, float scale,
                                          int lane) {
  v8f acc[4][2];
#pragma unroll
  for (int mt = 0; mt < 4; ++mt) { acc[mt][0] = {}; acc[mt][1] = {}; }
  for (int k0 = 0; k0 < D_MODEL; k0 += 32) {
    v16h a[4], b[2];
#pragma unroll
    for (int mt = 0; mt < 4; ++mt) a[mt] = a_frag_l(srcA, D_MODEL, mt * 16, k0, lane);
#pragma unroll
    for (int nt = 0; nt < 2; ++nt) b[nt] = b_frag(wT, D_MODEL, n0 + nt * 16, k0, lane);
#pragma unroll
    for (int mt = 0; mt < 4; ++mt)
#pragma unroll
      for (int nt = 0; nt < 2; ++nt)
        acc[mt][nt] = wmma_f16(a[mt], b[nt], acc[mt][nt]);
  }
#pragma unroll
  for (int mt = 0; mt < 4; ++mt)
#pragma unroll
    for (int nt = 0; nt < 2; ++nt) {
      if (TRANSPOSED) store_tile_f16_T(dst, N_TOK, mt * 16, n0 + nt * 16, acc[mt][nt], lane);
      else            store_tile_f16(dst, D_MODEL, mt * 16, n0 + nt * 16, acc[mt][nt], scale, lane);
    }
}

// ---- Kernel 0: convert+transpose weights fp32[k][n] -> f16 Wt[n][k] in ws ----
__global__ __launch_bounds__(256) void prep_weights(
    const float* __restrict__ wq, const float* __restrict__ wk,
    const float* __restrict__ wv, const float* __restrict__ wo,
    _Float16* __restrict__ wt) {
  int tid = blockIdx.x * 256 + threadIdx.x;       // 0..262143
  int m = tid >> 16;                              // matrix id
  int i = tid & 65535;
  int n = i >> 8, k = i & 255;
  const float* src = (m == 0) ? wq : (m == 1) ? wk : (m == 2) ? wv : wo;
  wt[m * 65536 + n * D_MODEL + k] = (_Float16)src[k * D_MODEL + n];
}

// ---- Kernel 1: one block per window ----
__global__ __launch_bounds__(256) void win_attn(
    const float* __restrict__ x, const float* __restrict__ cc,
    const _Float16* __restrict__ wt, const float* __restrict__ tab,
    float* __restrict__ out) {
  __shared__ _Float16 sBuf[4 * WIN_ELEMS];  // 128 KiB, 4 retimed 32 KiB regions
  __shared__ float    sTab[TAB_ELEMS];      // 7.2 KiB rel-pos bias table

  const int win  = blockIdx.x;
  const int wave = threadIdx.x >> 5;        // == head index
  const int lane = threadIdx.x & 31;
  const int nlo  = lane & 15;
  const int hb   = (lane >> 4) << 3;

  _Float16* sX  = sBuf;                     // R0: Xh,  later V^T [d][token]
  _Float16* sC  = sBuf + WIN_ELEMS;         // R1: Ch,  later P (waves 0-3)
  _Float16* sQ  = sBuf + 2 * WIN_ELEMS;     // R2: Q,   later P (waves 4-7)
  _Float16* sK  = sBuf + 3 * WIN_ELEMS;     // R3: K,   later O
  _Float16* sVt = sX;
  _Float16* sO  = sK;

  const float* xw = x  + (size_t)win * WIN_ELEMS;
  const float* cw = cc + (size_t)win * WIN_ELEMS;
  const _Float16* Wqt = wt;
  const _Float16* Wkt = wt + 65536;
  const _Float16* Wvt = wt + 131072;
  const _Float16* Wot = wt + 196608;
  const int n0 = wave * D_HEAD;

  // ---- stage x and c as f16 into LDS (each element converted exactly once) ----
  for (int t = threadIdx.x; t < TAB_ELEMS; t += 256) sTab[t] = tab[t];
  for (int i = threadIdx.x * 4; i < WIN_ELEMS; i += 1024) {
    v4f fx = *reinterpret_cast<const v4f*>(xw + i);
    v4f fc = *reinterpret_cast<const v4f*>(cw + i);
    v4h hx, hc;
#pragma unroll
    for (int t = 0; t < 4; ++t) { hx[t] = (_Float16)fx[t]; hc[t] = (_Float16)fc[t]; }
    *reinterpret_cast<v4h*>(sX + i) = hx;
    *reinterpret_cast<v4h*>(sC + i) = hc;
  }
  __syncthreads();   // sync0: staging complete

  // ---- Q projection (reads Xh/R0) ----
  proj_head<false>(sX, Wqt, sQ, n0, 0.17677669529663687f /* 1/sqrt(32) */, lane);
  __syncthreads();   // sync1: all Xh reads done -> R0 reusable as V^T

  // ---- K and V projections (read Ch/R1; V writes transposed into R0) ----
  proj_head<false>(sC, Wkt, sK, n0, 1.0f, lane);
  proj_head<true >(sC, Wvt, sVt, n0, 1.0f, lane);

  // ---- S = Q_h K_h^T (own-head slices: no cross-wave dependency) ----
  v8f s[4][4];
  {
    v16h aq[4], bk[4];
#pragma unroll
    for (int mt = 0; mt < 4; ++mt) aq[mt] = a_frag_l(sQ, D_MODEL, mt * 16, n0, lane);
#pragma unroll
    for (int nt = 0; nt < 4; ++nt) bk[nt] = b_frag(sK, D_MODEL, nt * 16, n0, lane);
#pragma unroll
    for (int mt = 0; mt < 4; ++mt)
#pragma unroll
      for (int nt = 0; nt < 4; ++nt) {
        v8f z = {};
        s[mt][nt] = wmma_f16(aq[mt], bk[nt], z);
      }
  }
  // relative position bias
#pragma unroll
  for (int mt = 0; mt < 4; ++mt)
#pragma unroll
    for (int nt = 0; nt < 4; ++nt)
#pragma unroll
      for (int r = 0; r < 8; ++r) {
        int gi = mt * 16 + hb + r;     // query token in 8x8 window
        int gj = nt * 16 + nlo;        // key token
        int idx = ((gi >> 3) - (gj >> 3) + 7) * 15 + ((gi & 7) - (gj & 7) + 7);
        s[mt][nt][r] += sTab[idx * N_HEAD + wave];
      }
  __syncthreads();   // sync2: all Ch/Q reads done -> R1+R2 reusable as P

  // ---- softmax over rows (C layout: lane halves hold disjoint row sets) ----
  _Float16* myP = sC + wave * (N_TOK * N_TOK);   // 8 KiB per wave in R1+R2
#pragma unroll
  for (int mt = 0; mt < 4; ++mt) {
    v8f rmx = s[mt][0];
#pragma unroll
    for (int nt = 1; nt < 4; ++nt)
#pragma unroll
      for (int r = 0; r < 8; ++r) rmx[r] = fmaxf(rmx[r], s[mt][nt][r]);
#pragma unroll
    for (int m = 1; m < 16; m <<= 1)
#pragma unroll
      for (int r = 0; r < 8; ++r)
        rmx[r] = fmaxf(rmx[r], __shfl_xor(rmx[r], m, 32));
    v8f rsum = {};
#pragma unroll
    for (int nt = 0; nt < 4; ++nt)
#pragma unroll
      for (int r = 0; r < 8; ++r) {
        float e = __expf(s[mt][nt][r] - rmx[r]);
        s[mt][nt][r] = e;
        rsum[r] += e;
      }
#pragma unroll
    for (int m = 1; m < 16; m <<= 1)
#pragma unroll
      for (int r = 0; r < 8; ++r) rsum[r] += __shfl_xor(rsum[r], m, 32);
#pragma unroll
    for (int r = 0; r < 8; ++r) rmx[r] = 1.0f / rsum[r];
#pragma unroll
    for (int nt = 0; nt < 4; ++nt)
#pragma unroll
      for (int r = 0; r < 8; ++r)
        myP[(mt * 16 + hb + r) * N_TOK + nt * 16 + nlo] =
            (_Float16)(s[mt][nt][r] * rmx[r]);
  }

  // ---- O_h = P (64x64) @ V_h (64x32); store into R3 (own-head columns) ----
  {
    v8f oacc[4][2];
#pragma unroll
    for (int mt = 0; mt < 4; ++mt) { oacc[mt][0] = {}; oacc[mt][1] = {}; }
#pragma unroll
    for (int ks = 0; ks < 2; ++ks) {
      v16h ap[4], bv[2];
#pragma unroll
      for (int mt = 0; mt < 4; ++mt) ap[mt] = a_frag_l(myP, N_TOK, mt * 16, ks * 32, lane);
#pragma unroll
      for (int nt = 0; nt < 2; ++nt) bv[nt] = b_frag(sVt, N_TOK, n0 + nt * 16, ks * 32, lane);
#pragma unroll
      for (int mt = 0; mt < 4; ++mt)
#pragma unroll
        for (int nt = 0; nt < 2; ++nt)
          oacc[mt][nt] = wmma_f16(ap[mt], bv[nt], oacc[mt][nt]);
    }
#pragma unroll
    for (int mt = 0; mt < 4; ++mt)
#pragma unroll
      for (int nt = 0; nt < 2; ++nt)
        store_tile_f16(sO, D_MODEL, mt * 16, n0 + nt * 16, oacc[mt][nt], 1.0f, lane);
  }
  __syncthreads();   // sync3: O complete before its cross-wave reads

  // ---- Out = O (64x256) @ Wo (256x256), fp32 to global ----
  {
    float* ow = out + (size_t)win * WIN_ELEMS;
    v8f acc[4][2];
#pragma unroll
    for (int mt = 0; mt < 4; ++mt) { acc[mt][0] = {}; acc[mt][1] = {}; }
    const int c0 = wave * 32;
    for (int k0 = 0; k0 < D_MODEL; k0 += 32) {
      v16h a[4], b[2];
#pragma unroll
      for (int mt = 0; mt < 4; ++mt) a[mt] = a_frag_l(sO, D_MODEL, mt * 16, k0, lane);
#pragma unroll
      for (int nt = 0; nt < 2; ++nt) b[nt] = b_frag(Wot, D_MODEL, c0 + nt * 16, k0, lane);
#pragma unroll
      for (int mt = 0; mt < 4; ++mt)
#pragma unroll
        for (int nt = 0; nt < 2; ++nt)
          acc[mt][nt] = wmma_f16(a[mt], b[nt], acc[mt][nt]);
    }
#pragma unroll
    for (int mt = 0; mt < 4; ++mt)
#pragma unroll
      for (int nt = 0; nt < 2; ++nt)
#pragma unroll
        for (int r = 0; r < 8; ++r)
          ow[(mt * 16 + hb + r) * D_MODEL + c0 + nt * 16 + nlo] = acc[mt][nt][r];
  }
}

extern "C" void kernel_launch(void* const* d_in, const int* in_sizes, int n_in,
                              void* d_out, int out_size, void* d_ws, size_t ws_size,
                              hipStream_t stream) {
  (void)in_sizes; (void)n_in; (void)out_size; (void)ws_size;
  const float* x   = (const float*)d_in[0];
  const float* c   = (const float*)d_in[1];
  const float* Wq  = (const float*)d_in[2];
  const float* Wk  = (const float*)d_in[3];
  const float* Wv  = (const float*)d_in[4];
  const float* Wo  = (const float*)d_in[5];
  const float* tab = (const float*)d_in[6];
  _Float16* wt = (_Float16*)d_ws;          // 4 * 256*256 f16 = 512 KiB

  prep_weights<<<1024, 256, 0, stream>>>(Wq, Wk, Wv, Wo, wt);
  win_attn<<<2048, 256, 0, stream>>>(x, c, wt, tab, (float*)d_out);
}
